// GCNClusterNet_23244363006576
// MI455X (gfx1250) — matrix operand order repaired and checked by reading, not compiled, not run on hardware
//
#include <hip/hip_runtime.h>
#include <stdint.h>

// ---------------------------------------------------------------------------
// GCN + soft k-means (ClusterNet) for MI455X / gfx1250, wave32.
// - Dense GEMMs via v_wmma_f32_16x16x32_bf16 (f32 accumulation), dual
//   accumulators to break the WMMA->WMMA RAW hazard chain.
// - LDS tiles staged with the Tensor Data Mover (tensor_load_to_lds +
//   s_wait_tensorcnt); TDM zero-fills OOB rows so tail tiles need no guards.
// - Fragments are contiguous ds_load_b128s (B kept K-major in LDS).
// - Edge scatter (~1.3 GB random traffic) dominates the roofline; it uses
//   feature-contiguous lanes + f32 global atomics.
// ---------------------------------------------------------------------------

#define NNODES 50000
#define NEDGES 800000
#define FIN    256
#define FHID   128
#define FOUT   64
#define KCLUS  50
#define KPAD   64
#define CTEMP  10.0f
#define NITER  10   // reference NUM_ITER (num_iter input is fixed to 10)

typedef __attribute__((ext_vector_type(16))) __bf16        v16bf;
typedef __attribute__((ext_vector_type(8)))  __bf16        v8bf;
typedef __attribute__((ext_vector_type(8)))  float         v8f;
typedef __attribute__((ext_vector_type(4)))  unsigned int  u32x4;
typedef __attribute__((ext_vector_type(8)))  int           i32x8;
typedef __attribute__((ext_vector_type(4)))  int           i32x4;

static inline int cb(long long n) { return (int)((n + 255) / 256); }

// --------------------- Tensor Data Mover descriptor ------------------------
// 2D tile load, bf16 elements (data_size=2B). D# bitfields per CDNA5 ISA §8.
// OOB rows (y >= t1) read as zero. 1D loads: d1=t1=1.
__device__ __forceinline__ void tdm_load_bf16(const void* gptr, unsigned lds_off,
                                              unsigned d0, unsigned d1,
                                              unsigned t0, unsigned t1,
                                              unsigned stride0) {
  unsigned long long ga = (unsigned long long)(uintptr_t)gptr;
  u32x4 g0;
  g0[0] = 1u;                                            // count=1 (user D#)
  g0[1] = lds_off;                                       // lds_addr
  g0[2] = (unsigned)(ga & 0xFFFFFFFFu);                  // global_addr[31:0]
  g0[3] = (unsigned)((ga >> 32) & 0x1FFFFFFu) | (2u << 30); // addr[56:32] | type=2
  i32x8 g1;
  g1[0] = (int)(1u << 16);                               // data_size=1 (2 bytes)
  g1[1] = (int)((t0 & 0xFFFFu) << 16);                   // tensor_dim0 lo
  g1[2] = (int)((t0 >> 16) | ((t1 & 0xFFFFu) << 16));    // tdim0 hi | tdim1 lo
  g1[3] = (int)((t1 >> 16) | (d0 << 16));                // tdim1 hi | tile_dim0
  g1[4] = (int)(d1 & 0xFFFFu);                           // tile_dim1 (tile_dim2=0)
  g1[5] = (int)stride0;                                  // tensor_dim0_stride lo
  g1[6] = 0;                                             // stride0 hi | stride1 lo
  g1[7] = 0;                                             // stride1 hi
  i32x4 z4 = {0, 0, 0, 0};
#if __clang_major__ >= 23
  i32x8 z8 = {0, 0, 0, 0, 0, 0, 0, 0};
  __builtin_amdgcn_tensor_load_to_lds(g0, g1, z4, z4, z8, 0);
#else
  __builtin_amdgcn_tensor_load_to_lds(g0, g1, z4, z4, 0);
#endif
}

// bf16 A-fragment (16x32 tile slice) from a row-major LDS row: two b128 runs.
__device__ __forceinline__ v16bf frag_a(const __bf16* row, int k0, int half) {
  v8bf lo = *(const v8bf*)(row + k0 + half * 8);
  v8bf hi = *(const v8bf*)(row + k0 + 16 + half * 8);
  return __builtin_shufflevector(lo, hi, 0, 1, 2, 3, 4, 5, 6, 7,
                                 8, 9, 10, 11, 12, 13, 14, 15);
}

// -------------------------- tiny utility kernels ---------------------------

__global__ void k_fill(float* p, float v, int n) {
  int i = blockIdx.x * 256 + threadIdx.x;
  if (i < n) p[i] = v;
}

__global__ void k_degree(const long long* __restrict__ ei, float* deg, int E) {
  int e = blockIdx.x * 256 + threadIdx.x;
  if (e < E) atomicAdd(&deg[(int)ei[(long long)E + e]], 1.0f);
}

__global__ void k_rsqrt_inplace(float* p, int n) {
  int i = blockIdx.x * 256 + threadIdx.x;
  if (i < n) p[i] = rsqrtf(p[i]);
}

__global__ void k_relu(float* p, long long n) {
  long long i = (long long)blockIdx.x * 256 + threadIdx.x;
  if (i < n) p[i] = fmaxf(p[i], 0.0f);
}

// f32 -> bf16 row-major copy
__global__ void k_cvt_bf16(const float* __restrict__ s, __bf16* __restrict__ d, long long n) {
  long long i = (long long)blockIdx.x * 256 + threadIdx.x;
  if (i < n) d[i] = (__bf16)s[i];
}

// f32 [K][NC] row-major -> bf16 [NC][K] (K-major, for contiguous B fragments)
__global__ void k_cvt_bf16_T(const float* __restrict__ s, __bf16* __restrict__ d, int K, int NC) {
  long long i = (long long)blockIdx.x * 256 + threadIdx.x;
  if (i < (long long)K * NC) {
    int k = (int)(i / NC), c = (int)(i % NC);
    d[(long long)c * K + k] = (__bf16)s[i];
  }
}

// ------------------------- bf16 WMMA dense GEMM ----------------------------
// C[M x NCOL] = A[M x K] * B[K x NCOL]; A bf16 row-major, BT bf16 [NCOL][K].
// 256 thr = 8 waves; wave = one 16x16 tile. A/B tiles staged by the TDM.
// Fragment layouts per CDNA5 ISA 7.12.2:
//   A lane(l): row=l&15, halves of K -> two contiguous 8-elem runs.
//   B lane(l): col=l&15, same K order as A -> two contiguous 8-elem runs.

template <int K, int NCOL>
__global__ void k_gemm_bf16(const __bf16* __restrict__ A, const __bf16* __restrict__ BT,
                            float* __restrict__ C, int M) {
  constexpr int COLT = NCOL / 16;
  constexpr int ROWT = 8 / COLT;
  constexpr int MB   = 16 * ROWT;
  extern __shared__ char smem[];
  __bf16* sA  = (__bf16*)smem;                         // [MB][K]
  __bf16* sBT = (__bf16*)(smem + (size_t)MB * K * 2);  // [NCOL][K]

  const int tid = threadIdx.x;
  const long long rowBase = (long long)blockIdx.x * MB;

  if (tid == 0) {
    // A tile: MB x K, zero-filled past row M-rowBase by TDM OOB handling.
    tdm_load_bf16(A + rowBase * K, (unsigned)(uintptr_t)sA,
                  K, MB, K, (unsigned)(M - rowBase), K);
    // BT: contiguous 1D block.
    tdm_load_bf16(BT, (unsigned)(uintptr_t)sBT,
                  NCOL * K, 1, NCOL * K, 1, NCOL * K);
    __builtin_amdgcn_s_wait_tensorcnt(0);
  }
  __syncthreads();

  const int w = tid >> 5, lane = tid & 31;
  const int rt = w / COLT, ct = w % COLT;
  const int half = lane >> 4, mn = lane & 15;
  const __bf16* sAr = sA  + (size_t)(rt * 16 + mn) * K;
  const __bf16* sBc = sBT + (size_t)(ct * 16 + mn) * K;
  const int col = ct * 16 + mn;

  // Dual accumulators: alternate K-chunks so back-to-back WMMAs are
  // independent (avoids the 5-slot WMMA->WMMA RAW hazard NOPs).
  v8f acc0 = {0.f, 0.f, 0.f, 0.f, 0.f, 0.f, 0.f, 0.f};
  v8f acc1 = {0.f, 0.f, 0.f, 0.f, 0.f, 0.f, 0.f, 0.f};
#pragma unroll
  for (int k0 = 0; k0 < K; k0 += 64) {
    v16bf a0 = frag_a(sAr, k0, half);
    v16bf b0 = frag_a(sBc, k0, half);
    acc0 = __builtin_amdgcn_wmma_f32_16x16x32_bf16(false, a0, false, b0,
                                                   (short)0, acc0, false, false);
    v16bf a1 = frag_a(sAr, k0 + 32, half);
    v16bf b1 = frag_a(sBc, k0 + 32, half);
    acc1 = __builtin_amdgcn_wmma_f32_16x16x32_bf16(false, a1, false, b1,
                                                   (short)0, acc1, false, false);
  }
  v8f acc = acc0 + acc1;

  float* Crow = C + (rowBase + rt * 16 + 8 * half) * NCOL + col;
  if (rowBase + MB <= M) {            // uniform fast path: full tile
#pragma unroll
    for (int r = 0; r < 8; ++r) Crow[(size_t)r * NCOL] = acc[r];
  } else {
#pragma unroll
    for (int r = 0; r < 8; ++r) {
      long long gr = rowBase + rt * 16 + r + 8 * half;
      if (gr < M) C[gr * NCOL + col] = acc[r];
    }
  }
}

// --------------------- GCN aggregation (scatter) ---------------------------

__global__ void k_agg_init(const float* __restrict__ H, const float* __restrict__ bias,
                           const float* __restrict__ dinv, float* __restrict__ out,
                           int n, int F) {
  long long i = (long long)blockIdx.x * 256 + threadIdx.x;
  if (i >= (long long)n * F) return;
  int node = (int)(i / F), f = (int)(i % F);
  float di = dinv[node];
  out[i] = bias[f] + H[i] * di * di;   // bias + self-loop term
}

__global__ void k_agg_edges(const long long* __restrict__ ei, const float* __restrict__ H,
                            const float* __restrict__ dinv, float* __restrict__ out,
                            int E, int F) {
  long long t = (long long)blockIdx.x * 256 + threadIdx.x;
  if (t >= (long long)E * F) return;
  int e = (int)(t / F), f = (int)(t % F);
  int s = (int)ei[e];
  int d = (int)ei[(long long)E + e];
  float c = dinv[s] * dinv[d];
  atomicAdd(&out[(long long)d * F + f], H[(long long)s * F + f] * c);
}

// ------------- row log_softmax + L2 row-normalize (wave per row) -----------
// Also emits the bf16 copy used as WMMA A-operand by the clustering kernels.

__global__ void k_lsm_norm(float* __restrict__ emb, __bf16* __restrict__ embbf, int n) {
  int row = blockIdx.x * 8 + (threadIdx.x >> 5);
  int lane = threadIdx.x & 31;
  if (row >= n) return;
  float* p = emb + (long long)row * FOUT;
  float v0 = p[lane], v1 = p[lane + 32];
  float mx = fmaxf(v0, v1);
  for (int off = 16; off > 0; off >>= 1) mx = fmaxf(mx, __shfl_xor(mx, off, 32));
  float s = expf(v0 - mx) + expf(v1 - mx);
  for (int off = 16; off > 0; off >>= 1) s += __shfl_xor(s, off, 32);
  float lg = logf(s);
  float l0 = v0 - mx - lg, l1 = v1 - mx - lg;
  float sq = l0 * l0 + l1 * l1;
  for (int off = 16; off > 0; off >>= 1) sq += __shfl_xor(sq, off, 32);
  float inv = rsqrtf(sq);
  float d0 = l0 * inv, d1 = l1 * inv;
  p[lane] = d0;
  p[lane + 32] = d1;
  __bf16* q = embbf + (long long)row * FOUT;
  q[lane] = (__bf16)d0;
  q[lane + 32] = (__bf16)d1;
}

// ----------------------------- clustering ----------------------------------

__global__ void k_init_mu(const float* __restrict__ init, float* mu, __bf16* mubf) {
  int i = blockIdx.x * 256 + threadIdx.x;
  if (i < KPAD * FOUT) {
    int k = i >> 6;
    float v = (k < KCLUS) ? init[i] : 0.0f;
    mu[i] = v;
    mubf[i] = (__bf16)v;
  }
}

// mode==0: dist = data@mu.T via WMMA, softmax over k, accumulate CR/CM.
// mode==1: write raw dist[N][KPAD] and exit.
__global__ void k_cluster(const float* __restrict__ data, const __bf16* __restrict__ databf,
                          const __bf16* __restrict__ mubf,
                          float* gCR, float* gCM, float* dist_out, int n, int mode) {
  extern __shared__ char smem[];
  __bf16* sMu   = (__bf16*)smem;                          // [KPAD][FOUT] bf16 (8KB)
  __bf16* sDb   = (__bf16*)(smem + KPAD * FOUT * 2);      // [128][FOUT] bf16 (16KB)
  float*  sData = (float*)(smem + KPAD * FOUT * 2 + 128 * FOUT * 2); // [128][FOUT] (32KB)
  float*  sR    = sData + 128 * FOUT;                     // [128][KPAD] (32KB)

  const int tid = threadIdx.x;
  const int nb = blockIdx.x * 128;
  const bool fullTile = (nb + 128 <= n);

  if (tid == 0) {
    tdm_load_bf16(mubf, (unsigned)(uintptr_t)sMu,
                  KPAD * FOUT, 1, KPAD * FOUT, 1, KPAD * FOUT);
    tdm_load_bf16(databf + (long long)nb * FOUT, (unsigned)(uintptr_t)sDb,
                  FOUT, 128, FOUT, (unsigned)(n - nb), FOUT);   // OOB rows -> 0
    __builtin_amdgcn_s_wait_tensorcnt(0);
  }
  if (!mode) {
    for (int i = tid; i < 128 * FOUT; i += 256) {
      int r = i >> 6, gn = nb + r;
      sData[i] = (gn < n) ? data[(long long)gn * FOUT + (i & 63)] : 0.0f;
    }
  }
  __syncthreads();

  const int w = tid >> 5, lane = tid & 31;
  const int half = lane >> 4, mn = lane & 15;
  const __bf16* arow = sDb + (size_t)(w * 16 + mn) * FOUT;

#pragma unroll
  for (int ct = 0; ct < KPAD / 16; ++ct) {
    const int col = ct * 16 + mn;
    const __bf16* brow = sMu + (size_t)col * FOUT;  // B = mu.T, K-major = mu row
    // FOUT=64 -> exactly two independent WMMAs (no accumulator hazard).
    v8f acc0 = {0.f, 0.f, 0.f, 0.f, 0.f, 0.f, 0.f, 0.f};
    v8f acc1 = {0.f, 0.f, 0.f, 0.f, 0.f, 0.f, 0.f, 0.f};
    v16bf a0 = frag_a(arow, 0, half);
    v16bf b0 = frag_a(brow, 0, half);
    acc0 = __builtin_amdgcn_wmma_f32_16x16x32_bf16(false, a0, false, b0,
                                                   (short)0, acc0, false, false);
    v16bf a1 = frag_a(arow, 32, half);
    v16bf b1 = frag_a(brow, 32, half);
    acc1 = __builtin_amdgcn_wmma_f32_16x16x32_bf16(false, a1, false, b1,
                                                   (short)0, acc1, false, false);
    v8f acc = acc0 + acc1;

    if (mode) {
      if (fullTile) {   // uniform fast path (all but the last block)
        float* dr = dist_out + (long long)(nb + w * 16 + 8 * half) * KPAD + col;
#pragma unroll
        for (int r = 0; r < 8; ++r) dr[(size_t)r * KPAD] = acc[r];
      } else {
#pragma unroll
        for (int r = 0; r < 8; ++r) {
          int gr = nb + w * 16 + r + 8 * half;
          if (gr < n) dist_out[(long long)gr * KPAD + col] = acc[r];
        }
      }
    } else {
#pragma unroll
      for (int r = 0; r < 8; ++r) sR[(w * 16 + r + 8 * half) * KPAD + col] = acc[r];
    }
  }
  if (mode) return;
  __syncthreads();

  // per-node softmax over the 50 valid clusters (temp = 10)
  if (tid < 128) {
    int gn = nb + tid;
    float* rr = sR + tid * KPAD;
    if (gn < n) {
      float mx = -1e30f;
      for (int k = 0; k < KCLUS; ++k) mx = fmaxf(mx, CTEMP * rr[k]);
      float sum = 0.f;
      for (int k = 0; k < KCLUS; ++k) { float e = expf(CTEMP * rr[k] - mx); rr[k] = e; sum += e; }
      float inv = 1.f / sum;
      for (int k = 0; k < KCLUS; ++k) rr[k] *= inv;
    } else {
      for (int k = 0; k < KCLUS; ++k) rr[k] = 0.0f;
    }
  }
  __syncthreads();

  if (tid < KCLUS) {               // cluster_r partials
    float s = 0.f;
    for (int r = 0; r < 128; ++r) s += sR[r * KPAD + tid];
    atomicAdd(&gCR[tid], s);
  }
  for (int e = tid; e < KCLUS * FOUT; e += 256) {   // r.T @ data partials
    int k = e >> 6, d = e & 63;
    float s = 0.f;
    for (int r = 0; r < 128; ++r) s += sR[r * KPAD + k] * sData[r * FOUT + d];
    atomicAdd(&gCM[k * FOUT + d], s);
  }
}

__global__ void k_mu_update(const float* __restrict__ gCR, const float* __restrict__ gCM,
                            float* mu, __bf16* mubf) {
  int i = blockIdx.x * 256 + threadIdx.x;
  if (i < KPAD * FOUT) {
    int k = i >> 6;
    float v = (k < KCLUS) ? gCM[i] / gCR[k] : 0.0f;
    mu[i] = v;
    mubf[i] = (__bf16)v;
  }
}

// ---------------- column softmax over N (axis=0) + output ------------------

__global__ void k_colsoftmax(const float* __restrict__ dist, float* cmax, float* csum, int n) {
  __shared__ float red[256];
  const int k = blockIdx.x, tid = threadIdx.x;
  float m = -1e30f;
  for (int i = tid; i < n; i += 256) m = fmaxf(m, CTEMP * dist[(long long)i * KPAD + k]);
  red[tid] = m; __syncthreads();
  for (int s = 128; s > 0; s >>= 1) { if (tid < s) red[tid] = fmaxf(red[tid], red[tid + s]); __syncthreads(); }
  m = red[0]; __syncthreads();
  float su = 0.f;
  for (int i = tid; i < n; i += 256) su += expf(CTEMP * dist[(long long)i * KPAD + k] - m);
  red[tid] = su; __syncthreads();
  for (int s = 128; s > 0; s >>= 1) { if (tid < s) red[tid] += red[tid + s]; __syncthreads(); }
  if (tid == 0) { cmax[k] = m; csum[k] = red[0]; }
}

__global__ void k_out_phase1(const float* __restrict__ dist, const float* __restrict__ cmax,
                             const float* __restrict__ csum, float* __restrict__ out,
                             float* ssum, int n) {
  __shared__ float sm[KCLUS], ss[KCLUS], red[256];
  const int tid = threadIdx.x;
  if (tid < KCLUS) { sm[tid] = cmax[tid]; ss[tid] = csum[tid]; }
  __syncthreads();
  int i = blockIdx.x * 256 + tid;
  float y = 0.f;
  if (i < n) {
    const float* dr = dist + (long long)i * KPAD;
    float o = 0.f;
    for (int k = 0; k < KCLUS; ++k) o += expf(CTEMP * dr[k] - sm[k]) / ss[k];
    y = 2.f * (1.f / (1.f + expf(-4.f * o)) - 0.5f);
    out[i] = y;
  }
  red[tid] = y; __syncthreads();
  for (int s = 128; s > 0; s >>= 1) { if (tid < s) red[tid] += red[tid + s]; __syncthreads(); }
  if (tid == 0) atomicAdd(ssum, red[0]);
}

__global__ void k_out_phase2(float* out, const float* __restrict__ ssum, int n) {
  int i = blockIdx.x * 256 + threadIdx.x;
  if (i < n) {
    float s = *ssum;
    if (s > (float)KCLUS) out[i] *= (float)KCLUS / s;
  }
}

// ------------------------------ driver -------------------------------------

extern "C" void kernel_launch(void* const* d_in, const int* in_sizes, int n_in,
                              void* d_out, int out_size, void* d_ws, size_t ws_size,
                              hipStream_t stream) {
  (void)in_sizes; (void)n_in; (void)out_size; (void)ws_size;
  const float*     x    = (const float*)d_in[0];
  const long long* ei   = (const long long*)d_in[1];   // int64 [2][E]
  // d_in[2] = num_iter (device scalar; reference fixes it at 10 -> hardcoded,
  // since graph capture forbids a synchronous readback).
  const float*     W1   = (const float*)d_in[3];
  const float*     b1   = (const float*)d_in[4];
  const float*     W2   = (const float*)d_in[5];
  const float*     b2   = (const float*)d_in[6];
  const float*     init = (const float*)d_in[7];
  float*           out  = (float*)d_out;

  char* ws = (char*)d_ws;
  size_t off = 0;
  auto alloc = [&](size_t bytes) { void* p = ws + off; off = (off + bytes + 255) & ~(size_t)255; return p; };
  float*  DINV  = (float*)alloc((size_t)NNODES * 4);
  float*  H1    = (float*)alloc((size_t)NNODES * FHID * 4);
  float*  H1A   = (float*)alloc((size_t)NNODES * FHID * 4);
  float*  H2    = (float*)alloc((size_t)NNODES * FOUT * 4);
  float*  EMB   = (float*)alloc((size_t)NNODES * FOUT * 4);
  __bf16* XBF   = (__bf16*)alloc((size_t)NNODES * FIN * 2);
  __bf16* H1ABF = (__bf16*)alloc((size_t)NNODES * FHID * 2);
  __bf16* EMBBF = (__bf16*)alloc((size_t)NNODES * FOUT * 2);
  __bf16* W1T   = (__bf16*)alloc((size_t)FIN * FHID * 2);   // [FHID][FIN]
  __bf16* W2T   = (__bf16*)alloc((size_t)FHID * FOUT * 2);  // [FOUT][FHID]
  float*  MU    = (float*)alloc((size_t)KPAD * FOUT * 4);
  __bf16* MUBF  = (__bf16*)alloc((size_t)KPAD * FOUT * 2);
  float*  CR    = (float*)alloc((size_t)KPAD * 4);
  float*  CM    = (float*)alloc((size_t)KPAD * FOUT * 4);
  float*  CMAX  = (float*)alloc((size_t)KPAD * 4);
  float*  CSUM  = (float*)alloc((size_t)KPAD * 4);
  float*  SSUM  = (float*)alloc(256);
  float*  DIST  = H1;  // H1 dead after layer-2 GEMM; reuse as dist [N][KPAD]

  // degree / sym-norm (self loop => deg starts at 1)
  k_fill<<<cb(NNODES), 256, 0, stream>>>(DINV, 1.0f, NNODES);
  k_degree<<<cb(NEDGES), 256, 0, stream>>>(ei, DINV, NEDGES);
  k_rsqrt_inplace<<<cb(NNODES), 256, 0, stream>>>(DINV, NNODES);

  // bf16 operand prep (weights transposed to K-major for B fragments)
  k_cvt_bf16<<<cb((long long)NNODES * FIN), 256, 0, stream>>>(x, XBF, (long long)NNODES * FIN);
  k_cvt_bf16_T<<<cb((long long)FIN * FHID), 256, 0, stream>>>(W1, W1T, FIN, FHID);
  k_cvt_bf16_T<<<cb((long long)FHID * FOUT), 256, 0, stream>>>(W2, W2T, FHID, FOUT);

  // layer 1: H1 = x @ W1 ; aggregate ; ReLU
  k_gemm_bf16<FIN, FHID><<<(NNODES + 15) / 16, 256,
      ((size_t)16 * FIN + (size_t)FHID * FIN) * 2, stream>>>(XBF, W1T, H1, NNODES);
  k_agg_init<<<cb((long long)NNODES * FHID), 256, 0, stream>>>(H1, b1, DINV, H1A, NNODES, FHID);
  k_agg_edges<<<cb((long long)NEDGES * FHID), 256, 0, stream>>>(ei, H1, DINV, H1A, NEDGES, FHID);
  k_relu<<<cb((long long)NNODES * FHID), 256, 0, stream>>>(H1A, (long long)NNODES * FHID);
  k_cvt_bf16<<<cb((long long)NNODES * FHID), 256, 0, stream>>>(H1A, H1ABF, (long long)NNODES * FHID);

  // layer 2: H2 = relu(H1A) @ W2 ; aggregate ; log_softmax + L2 normalize
  k_gemm_bf16<FHID, FOUT><<<(NNODES + 31) / 32, 256,
      ((size_t)32 * FHID + (size_t)FOUT * FHID) * 2, stream>>>(H1ABF, W2T, H2, NNODES);
  k_agg_init<<<cb((long long)NNODES * FOUT), 256, 0, stream>>>(H2, b2, DINV, EMB, NNODES, FOUT);
  k_agg_edges<<<cb((long long)NEDGES * FOUT), 256, 0, stream>>>(ei, H2, DINV, EMB, NEDGES, FOUT);
  k_lsm_norm<<<(NNODES + 7) / 8, 256, 0, stream>>>(EMB, EMBBF, NNODES);

  // soft k-means: 10 scan steps + 1 extra, then final dist
  k_init_mu<<<(KPAD * FOUT + 255) / 256, 256, 0, stream>>>(init, MU, MUBF);
  const size_t smC = (size_t)KPAD * FOUT * 2 + (size_t)128 * FOUT * 2
                   + (size_t)128 * FOUT * 4 + (size_t)128 * KPAD * 4;
  const int cblocks = (NNODES + 127) / 128;
  for (int it = 0; it < NITER + 1; ++it) {
    k_fill<<<1, 256, 0, stream>>>(CR, 0.0f, KPAD);
    k_fill<<<(KPAD * FOUT + 255) / 256, 256, 0, stream>>>(CM, 0.0f, KPAD * FOUT);
    k_cluster<<<cblocks, 256, smC, stream>>>(EMB, EMBBF, MUBF, CR, CM, nullptr, NNODES, 0);
    k_mu_update<<<(KPAD * FOUT + 255) / 256, 256, 0, stream>>>(CR, CM, MU, MUBF);
  }
  k_cluster<<<cblocks, 256, smC, stream>>>(EMB, EMBBF, MUBF, CR, CM, DIST, NNODES, 1);

  // out = softmax(10*dist, axis=0).sum(axis=1) -> sigmoid gate -> rescale
  k_colsoftmax<<<KCLUS, 256, 0, stream>>>(DIST, CMAX, CSUM, NNODES);
  k_fill<<<1, 256, 0, stream>>>(SSUM, 0.0f, 1);
  k_out_phase1<<<cb(NNODES), 256, 0, stream>>>(DIST, CMAX, CSUM, out, SSUM, NNODES);
  k_out_phase2<<<cb(NNODES), 256, 0, stream>>>(out, SSUM, NNODES);
}
